// StructuredFeatureExtraction_65592740544973
// MI455X (gfx1250) — matrix operand order repaired and checked by reading, not compiled, not run on hardware
//
#include <hip/hip_runtime.h>

// MI455X / gfx1250. B=256, P=361, F=20 -> 92416 tokens = 722 blocks x 128 tokens.
// Block = 256 threads = 8 wave32 waves; each wave owns a 16-token tile.
// All GEMM work via v_wmma_f32_16x16x32_f16 (f32 accum). LDS holds weights and
// activations pre-swizzled into WMMA fragment layout so every fragment load is
// two ds_load_b128 (no per-element gathers).

#define FBINS 20

typedef __attribute__((ext_vector_type(16))) _Float16 v16h;
typedef __attribute__((ext_vector_type(8)))  float    v8f;
typedef __attribute__((ext_vector_type(2)))  float    f32x2;

struct Params {
  const float* x;
  const float *rw1,*rb1,*rw2,*rb2,*rw3,*rb3;
  const float *iw1,*ib1,*iw2,*ib2,*iw3,*ib3;
  const float *fw1,*fb1,*fw2,*fb2;
  const float *freq_imp,*pw1,*pb1,*pw2,*pb2,*pw3,*pb3;
  const float *gw1,*gb1,*gw2,*gb2;
  float* out;
};

// exact GELU (matches jax.nn.gelu(approximate=False))
__device__ __forceinline__ float gelu_f(float v) {
  return 0.5f * v * (1.0f + erff(v * 0.70710678118654752440f));
}

__device__ __forceinline__ v8f wmma16(v16h a, v16h b, v8f c) {
  return __builtin_amdgcn_wmma_f32_16x16x32_f16(false, a, false, b, (short)0, c,
                                                false, false);
}

// Fragment load: per-lane 16 contiguous f16 (32 B) -> 2x ds_load_b128.
__device__ __forceinline__ v16h ldfrag(const _Float16* fr, int lane) {
  return *(const v16h*)(fr + lane * 16);
}

// Stage a row-major f32 weight W[K][N] into f16 WMMA B-fragment layout.
// Fragment (kc,nb) = 512 halves at (kc*(N/16)+nb)*512; within a fragment,
// lane (((kk>>3)&1) ? 16:0)+(n&15) holds K row kk (idx = (kk&7)+((kk>>4)<<3)) of column n.
template <int K, int N>
__device__ __forceinline__ void stage_frag(_Float16* dst, const float* src, int tid) {
  const int NB = N / 16;
  for (int e2 = tid; e2 < (K * N) / 2; e2 += 256) {
    f32x2 v = *(const f32x2*)(src + e2 * 2);
    int e  = e2 * 2;
    int k  = e / N;               // N is a power of two -> shift
    int n0 = e - k * N;
    int kk = k & 31;
#pragma unroll
    for (int t = 0; t < 2; ++t) {
      int n  = n0 + t;
      int fi = (k >> 5) * NB + (n >> 4);
      int lp = (((kk >> 3) & 1) << 4) + (n & 15);
      dst[fi * 512 + lp * 16 + (kk & 7) + ((kk >> 4) << 3)] = (_Float16)v[t];
    }
  }
}

// Store a C fragment (+bias, optional GELU) into an LDS A-fragment slot at
// K-columns col0..col0+15. A layout: lanes 0-15 hold row m, K 0-7 (idx 0-7)
// and K 16-23 (idx 8-15); lanes 16-31 hold K 8-15 / 24-31.
__device__ __forceinline__ void store_cfrag_A(_Float16* slot, int col0, v8f c,
                                              float bias, bool act, int lane) {
  int n   = lane & 15;
  int mb  = (lane >> 4) * 8;
  int kk  = col0 + n;
  int lh  = (kk >> 3) & 1;
  int idx = (kk & 7) + ((kk >> 4) << 3);
#pragma unroll
  for (int r = 0; r < 8; ++r) {
    float v = c[r] + bias;
    if (act) v = gelu_f(v);
    slot[(mb + r + (lh << 4)) * 16 + idx] = (_Float16)v;
  }
}

// bbuf layout (floats): 0 rb2 | 32 ib2 | 64 rb3 | 96 ib3 | 128 fb1 | 160 fb2
//                       192 rw1 | 208 rb1 | 224 iw1 | 240 ib1
__global__ __launch_bounds__(256) void sfe_gfx1250_kernel(Params p) {
  __shared__ _Float16 wfrag[8192];       // 16 KB: weight fragments (per phase)
  __shared__ float    bbuf[256];         //  1 KB: per-freq biases + layer1 vecs
  __shared__ _Float16 aslot[8][4][512];  // 32 KB: per-wave activation A-slots
  __shared__ _Float16 magf[8][512];      //  8 KB: per-wave mag A-slot (K pad 32)

  const int tid  = threadIdx.x;
  const int wave = tid >> 5;             // wave32
  const int lane = tid & 31;
  const int n15  = lane & 15;
  const int tokbase = blockIdx.x * 128 + wave * 16;

  _Float16* s0 = aslot[wave][0];
  _Float16* s1 = aslot[wave][1];
  _Float16* s2 = aslot[wave][2];
  _Float16* s3 = aslot[wave][3];
  _Float16* mg = magf[wave];

  // One-time zero: mag slots (K 20..31 pad) and wfrag (so K=16 weight fragments
  // keep idx 8-15 (K 16-31) == 0 in every lane across the whole f-loop; staging never touches them).
  for (int i = tid; i < 8 * 512; i += 256) (&magf[0][0])[i] = (_Float16)0.0f;
  for (int i = tid; i < 8192;   i += 256) wfrag[i] = (_Float16)0.0f;

  // softmax(freq_imp) constants (uniform scalar work)
  float fmax_ = -1e30f;
  for (int f = 0; f < FBINS; ++f) fmax_ = fmaxf(fmax_, p.freq_imp[f]);
  float fsum = 0.0f;
  for (int f = 0; f < FBINS; ++f) fsum += expf(p.freq_imp[f] - fmax_);
  const float rsum = 1.0f / fsum;

  v8f acc0 = {}, acc1 = {};   // running sum of complex_features (cols 0-15/16-31)

  for (int f = 0; f < FBINS; ++f) {
    __syncthreads();  // previous iteration's wfrag/bbuf reads done (covers init)

    // ---- stage per-freq weights into fragment layout (coalesced float2) ----
    stage_frag<16, 32>(wfrag +  0 * 512, p.rw2 + f * 512,  tid);  // frags 0-1
    stage_frag<16, 32>(wfrag +  2 * 512, p.iw2 + f * 512,  tid);  // frags 2-3
    stage_frag<32, 32>(wfrag +  4 * 512, p.rw3 + f * 1024, tid);  // frags 4-5
    stage_frag<32, 32>(wfrag +  6 * 512, p.iw3 + f * 1024, tid);  // frags 6-7
    stage_frag<64, 32>(wfrag +  8 * 512, p.fw1 + f * 2048, tid);  // frags 8-11
    stage_frag<32, 32>(wfrag + 12 * 512, p.fw2 + f * 1024, tid);  // frags 12-13
    {
      int j32 = f * 32 + (tid & 31);
      int j16 = f * 16 + (tid & 15);
      float v;
      if (tid < 192) {
        switch (tid >> 5) {
          case 0:  v = p.rb2[j32]; break;
          case 1:  v = p.ib2[j32]; break;
          case 2:  v = p.rb3[j32]; break;
          case 3:  v = p.ib3[j32]; break;
          case 4:  v = p.fb1[j32]; break;
          default: v = p.fb2[j32]; break;
        }
        bbuf[tid] = v;
      } else {
        switch ((tid - 192) >> 4) {
          case 0:  v = p.rw1[j16]; break;
          case 1:  v = p.rb1[j16]; break;
          case 2:  v = p.iw1[j16]; break;
          default: v = p.ib1[j16]; break;
        }
        bbuf[tid] = v;
      }
    }
    if (tid == 0 && f + 1 < FBINS) {     // gfx1250 global_prefetch_b8
      __builtin_prefetch(p.fw1 + (f + 1) * 2048, 0, 0);
      __builtin_prefetch(p.rw3 + (f + 1) * 1024, 0, 0);
    }
    __syncthreads();

    // ---- layer 1 (elementwise) + mag, written straight in A-frag layout ----
    {
      int m  = n15;
      int jh = (lane >> 4) * 8;          // this lane covers j = jh..jh+7
      f32x2 ri = *(const f32x2*)(p.x + (tokbase + m) * (FBINS * 2) + f * 2);
      float re = ri[0], im = ri[1];
      v16h hr, hi;
#pragma unroll
      for (int jj = 0; jj < 8; ++jj) {
        int j = jh + jj;
        hr[jj] = (_Float16)gelu_f(re * bbuf[192 + j] + bbuf[208 + j]);
        hi[jj] = (_Float16)gelu_f(im * bbuf[224 + j] + bbuf[240 + j]);
        hr[8 + jj] = (_Float16)0.0f;     // K 16..31 zero pad
        hi[8 + jj] = (_Float16)0.0f;
      }
      *(v16h*)(s0 + lane * 16) = hr;     // h1r slot, one 32B store
      *(v16h*)(s1 + lane * 16) = hi;     // h1i slot
      if (lane < 16) {
        float wf  = expf(p.freq_imp[f] - fmax_) * rsum;
        float mag = sqrtf(re * re + im * im + 1e-8f);
        int lh = (f >> 3) & 1;
        int idx = (f & 7) + ((f >> 4) << 3);
        mg[(m + (lh << 4)) * 16 + idx] = (_Float16)(mag * wf);
      }
    }

    // ---- L2 real: h1r @ rw2 -> gelu -> h2r (s2) ----
    {
      v16h a = ldfrag(s0, lane);
#pragma unroll
      for (int nb = 0; nb < 2; ++nb) {
        v8f c = {};
        c = wmma16(a, ldfrag(wfrag + (0 + nb) * 512, lane), c);
        store_cfrag_A(s2, nb * 16, c, bbuf[0 + nb * 16 + n15], true, lane);
      }
    }
    // ---- L2 imag -> h2i (s3) ----
    {
      v16h a = ldfrag(s1, lane);
#pragma unroll
      for (int nb = 0; nb < 2; ++nb) {
        v8f c = {};
        c = wmma16(a, ldfrag(wfrag + (2 + nb) * 512, lane), c);
        store_cfrag_A(s3, nb * 16, c, bbuf[32 + nb * 16 + n15], true, lane);
      }
    }
    // ---- L3 real: h2r @ rw3 -> ro (s0, no act) ----
    {
      v16h a = ldfrag(s2, lane);
#pragma unroll
      for (int nb = 0; nb < 2; ++nb) {
        v8f c = {};
        c = wmma16(a, ldfrag(wfrag + (4 + nb) * 512, lane), c);
        store_cfrag_A(s0, nb * 16, c, bbuf[64 + nb * 16 + n15], false, lane);
      }
    }
    // ---- L3 imag -> io (s1) ----
    {
      v16h a = ldfrag(s3, lane);
#pragma unroll
      for (int nb = 0; nb < 2; ++nb) {
        v8f c = {};
        c = wmma16(a, ldfrag(wfrag + (6 + nb) * 512, lane), c);
        store_cfrag_A(s1, nb * 16, c, bbuf[96 + nb * 16 + n15], false, lane);
      }
    }
    // ---- F1: [ro|io](K=64) @ fw1 -> gelu -> h (s2) ----
    {
      v16h a0 = ldfrag(s0, lane);
      v16h a1 = ldfrag(s1, lane);
#pragma unroll
      for (int nb = 0; nb < 2; ++nb) {
        v8f c = {};
        c = wmma16(a0, ldfrag(wfrag + (8  + nb) * 512, lane), c);  // rows 0-31
        c = wmma16(a1, ldfrag(wfrag + (10 + nb) * 512, lane), c);  // rows 32-63
        store_cfrag_A(s2, nb * 16, c, bbuf[128 + nb * 16 + n15], true, lane);
      }
    }
    // ---- F2: h @ fw2 + fb2 -> accumulate mean in registers ----
    {
      v16h a = ldfrag(s2, lane);
      float b0 = bbuf[160 + n15];
      float b1 = bbuf[176 + n15];
      v8f t0 = {}, t1 = {};
      t0 = wmma16(a, ldfrag(wfrag + 12 * 512, lane), t0);
      t1 = wmma16(a, ldfrag(wfrag + 13 * 512, lane), t1);
#pragma unroll
      for (int r = 0; r < 8; ++r) { acc0[r] += t0[r] + b0; acc1[r] += t1[r] + b1; }
    }
  } // freq loop

#pragma unroll
  for (int r = 0; r < 8; ++r) { acc0[r] *= (1.0f / FBINS); acc1[r] *= (1.0f / FBINS); }

  // ==== tail phase 1: pw1/pw2/pw3 fragments ====
  __syncthreads();                                   // f-loop wfrag reads done
  for (int i = tid; i < 2048; i += 256) wfrag[i] = (_Float16)0.0f;  // pw1 K pad
  __syncthreads();
  stage_frag<20, 64>(wfrag,            p.pw1, tid);  // frags 0-3   (K20 padded)
  stage_frag<64, 64>(wfrag +  4 * 512, p.pw2, tid);  // frags 4-11  (kc*4+nb)
  stage_frag<64, 32>(wfrag + 12 * 512, p.pw3, tid);  // frags 12-15 (kc*2+nb)
  __syncthreads();

  // P1: mag(16x20 pad32) @ pw1 -> gelu -> fh1 (s0,s1)
  {
    v16h am = ldfrag(mg, lane);
#pragma unroll
    for (int nb = 0; nb < 4; ++nb) {
      v8f c = {};
      c = wmma16(am, ldfrag(wfrag + nb * 512, lane), c);
      _Float16* dst = (nb < 2) ? s0 : s1;
      store_cfrag_A(dst, (nb & 1) * 16, c, p.pb1[nb * 16 + n15], true, lane);
    }
  }
  // P2: fh1 @ pw2 -> gelu -> fh2 (s2,s3)
  {
    v16h a0 = ldfrag(s0, lane);
    v16h a1 = ldfrag(s1, lane);
#pragma unroll
    for (int nb = 0; nb < 4; ++nb) {
      v8f c = {};
      c = wmma16(a0, ldfrag(wfrag + (4 + nb) * 512, lane), c);
      c = wmma16(a1, ldfrag(wfrag + (8 + nb) * 512, lane), c);
      _Float16* dst = (nb < 2) ? s2 : s3;
      store_cfrag_A(dst, (nb & 1) * 16, c, p.pb2[nb * 16 + n15], true, lane);
    }
  }
  // P3: fh2 @ pw3 -> ff (s0) ; aggregated mean -> (s1)
  {
    v16h a0 = ldfrag(s2, lane);
    v16h a1 = ldfrag(s3, lane);
#pragma unroll
    for (int nb = 0; nb < 2; ++nb) {
      v8f c = {};
      c = wmma16(a0, ldfrag(wfrag + (12 + nb) * 512, lane), c);
      c = wmma16(a1, ldfrag(wfrag + (14 + nb) * 512, lane), c);
      store_cfrag_A(s0, nb * 16, c, p.pb3[nb * 16 + n15], false, lane);
    }
    store_cfrag_A(s1, 0,  acc0, 0.0f, false, lane);
    store_cfrag_A(s1, 16, acc1, 0.0f, false, lane);
  }
  // u = [ff | aggregated] kept in registers across restaging
  v16h u0 = ldfrag(s0, lane);
  v16h u1 = ldfrag(s1, lane);

  // ==== tail phase 2: gw1 ====
  __syncthreads();
  stage_frag<64, 128>(wfrag, p.gw1, tid);            // frags kc*8+nb (full)
  __syncthreads();
  // G1: u @ gw1 -> gelu -> g1(16x128) in s0..s3
  {
#pragma unroll
    for (int nb = 0; nb < 8; ++nb) {
      v8f c = {};
      c = wmma16(u0, ldfrag(wfrag + nb * 512, lane), c);
      c = wmma16(u1, ldfrag(wfrag + (8 + nb) * 512, lane), c);
      _Float16* dst = aslot[wave][nb >> 1];
      store_cfrag_A(dst, (nb & 1) * 16, c, p.gb1[nb * 16 + n15], true, lane);
    }
  }

  // ==== tail phase 3: gw2 ====
  __syncthreads();
  stage_frag<128, 64>(wfrag, p.gw2, tid);            // frags kc*4+nb (full)
  __syncthreads();
  // G2: g1 @ gw2 + gb2 -> out (B*P x 64 f32)
  {
    v16h a0 = ldfrag(s0, lane);
    v16h a1 = ldfrag(s1, lane);
    v16h a2 = ldfrag(s2, lane);
    v16h a3 = ldfrag(s3, lane);
    int mb_ = (lane >> 4) * 8;
#pragma unroll
    for (int nb = 0; nb < 4; ++nb) {
      v8f c = {};
      c = wmma16(a0, ldfrag(wfrag + (0  + nb) * 512, lane), c);
      c = wmma16(a1, ldfrag(wfrag + (4  + nb) * 512, lane), c);
      c = wmma16(a2, ldfrag(wfrag + (8  + nb) * 512, lane), c);
      c = wmma16(a3, ldfrag(wfrag + (12 + nb) * 512, lane), c);
      float bias = p.gb2[nb * 16 + n15];
#pragma unroll
      for (int r = 0; r < 8; ++r)
        p.out[(tokbase + mb_ + r) * 64 + nb * 16 + n15] = c[r] + bias;
    }
  }
}

extern "C" void kernel_launch(void* const* d_in, const int* in_sizes, int n_in,
                              void* d_out, int out_size, void* d_ws, size_t ws_size,
                              hipStream_t stream) {
  Params p;
  p.x        = (const float*)d_in[0];
  p.rw1      = (const float*)d_in[1];  p.rb1 = (const float*)d_in[2];
  p.rw2      = (const float*)d_in[3];  p.rb2 = (const float*)d_in[4];
  p.rw3      = (const float*)d_in[5];  p.rb3 = (const float*)d_in[6];
  p.iw1      = (const float*)d_in[7];  p.ib1 = (const float*)d_in[8];
  p.iw2      = (const float*)d_in[9];  p.ib2 = (const float*)d_in[10];
  p.iw3      = (const float*)d_in[11]; p.ib3 = (const float*)d_in[12];
  p.fw1      = (const float*)d_in[13]; p.fb1 = (const float*)d_in[14];
  p.fw2      = (const float*)d_in[15]; p.fb2 = (const float*)d_in[16];
  p.freq_imp = (const float*)d_in[17];
  p.pw1      = (const float*)d_in[18]; p.pb1 = (const float*)d_in[19];
  p.pw2      = (const float*)d_in[20]; p.pb2 = (const float*)d_in[21];
  p.pw3      = (const float*)d_in[22]; p.pb3 = (const float*)d_in[23];
  p.gw1      = (const float*)d_in[24]; p.gb1 = (const float*)d_in[25];
  p.gw2      = (const float*)d_in[26]; p.gb2 = (const float*)d_in[27];
  p.out      = (float*)d_out;

  int tokens = in_sizes[0] / (FBINS * 2);   // 92416
  int blocks = tokens / 128;                 // 722 (exact)
  hipLaunchKernelGGL(sfe_gfx1250_kernel, dim3(blocks), dim3(256), 0, stream, p);
}
